// MultiFrmVQBottleNeck_83124797047422
// MI455X (gfx1250) — matrix operand order repaired and checked by reading, not compiled, not run on hardware
//
#include <hip/hip_runtime.h>

typedef __attribute__((ext_vector_type(16))) __bf16 v16bf;
typedef __attribute__((ext_vector_type(8)))  __bf16 v8bf;
typedef __attribute__((ext_vector_type(8)))  float  v8f;

#define B_    32
#define FEAT_ 512
#define T_    1024
#define LAT_  288
#define G_    4
#define K_    1024
#define D_    144
#define DP_   160          // D padded to 5 chunks of 32 for bf16 WMMA
#define TC_   512          // T / CF
#define ROWS_ (B_ * TC_)   // 16384 VQ rows
#define NT1_  (LAT_ / 16)  // 18 l-tiles in proj1
#define KCH_  32           // k-chunk (WMMA K)
#define SEGS_ (LAT_ * (KCH_ * 2 / 8))   // 2304 8-byte segments per W1 chunk
#define SEGT_ (SEGS_ / 256)             // 9 async b64 ops per thread (uniform/wave)

static __device__ __forceinline__ v8f wmma_bf16(v16bf a, v16bf b, v8f c) {
    return __builtin_amdgcn_wmma_f32_16x16x32_bf16(false, a, false, b, (short)0, c,
                                                   false, false);
}

// ---- prep: codebooks -> bf16 padded [G][K][DP_], e2[G][K] -------------------
__global__ void k_prep_cb(const float* __restrict__ cb, __bf16* __restrict__ cbp,
                          float* __restrict__ e2) {
    int wave = (blockIdx.x * blockDim.x + threadIdx.x) >> 5;   // one wave per code
    int lane = threadIdx.x & 31;
    const float* src = cb + (size_t)wave * D_;
    __bf16* dst = cbp + (size_t)wave * DP_;
    float acc = 0.f;
    for (int d = lane; d < DP_; d += 32) {
        float v = (d < D_) ? src[d] : 0.f;
        acc += v * v;
        dst[d] = (__bf16)v;
    }
    for (int m = 1; m < 32; m <<= 1) acc += __shfl_xor(acc, m, 32);
    if (lane == 0) e2[wave] = acc;
}

// ---- prep: generic f32 -> bf16 convert (for W1 and W2) ----------------------
__global__ void k_cvt(const float* __restrict__ src, __bf16* __restrict__ dst, int n) {
    int i = blockIdx.x * blockDim.x + threadIdx.x;
    if (i < n) dst[i] = (__bf16)src[i];
}

// ---- GEMM1: z = x . W1^T, written as padded VQ rows vq[row][g*DP_+d] --------
// 8 waves/block, one 16-wide t-tile each, all 18 l-tile accumulators resident.
// W1 k-chunks (288x32 bf16 = 18KB) are double-buffered in LDS and filled with
// gfx1250 async load-to-LDS (ASYNCcnt); B fragments are software-pipelined one
// nt-tile ahead so ds latency hides behind WMMA.
__global__ void __launch_bounds__(256, 1)
k_proj1(const float* __restrict__ x, const __bf16* __restrict__ w1b,
        __bf16* __restrict__ vq) {
    __shared__ __align__(64) __bf16 smw[2][LAT_ * KCH_];   // 2 x 18KB

    int tid  = threadIdx.x;
    int lane = tid & 31;
    int tile = blockIdx.x * 8 + (tid >> 5);   // 8 t-tiles (same W1) per block
    int tt = tile % (T_ / 16);
    int b  = tile / (T_ / 16);
    int half = lane >> 4, ln = lane & 15;
    int t0 = tt * 16;

    const float* xrow = x + (size_t)b * FEAT_ * T_ + t0 + ln;  // A row m = ln

    // issue one async stage of W1 chunk `kc` into buffer `buf` (9 b64 ops/thread)
    auto stage = [&](int buf, int kc) {
        const char* gbase = (const char*)w1b + (size_t)kc * 2;
        unsigned lbase = (unsigned)(uintptr_t)&smw[buf][0];
        #pragma unroll
        for (int k = 0; k < SEGT_; ++k) {
            int s   = tid + k * 256;          // 8B segment id
            int row = s >> 3, cs = s & 7;
            const char* g = gbase + (size_t)row * (FEAT_ * 2) + cs * 8;
            unsigned    l = lbase + row * (KCH_ * 2) + cs * 8;
            asm volatile("global_load_async_to_lds_b64 %0, %1, off"
                         :: "v"(l), "v"(g) : "memory");
        }
    };

    v8f acc[NT1_];
    #pragma unroll
    for (int i = 0; i < NT1_; ++i) { v8f z = {}; acc[i] = z; }

    stage(0, 0);
    stage(1, KCH_);

    #pragma unroll 1
    for (int kci = 0; kci < FEAT_ / KCH_; ++kci) {
        int kc  = kci * KCH_;
        int cur = kci & 1;

        // my async ops for buffer `cur` done (next stage's 9 may stay in flight)
        if (kci < FEAT_ / KCH_ - 1)
            asm volatile("s_wait_asynccnt 0x9" ::: "memory");
        else
            asm volatile("s_wait_asynccnt 0x0" ::: "memory");
        __syncthreads();                       // whole buffer valid

        // batched strided f32 A-loads, then convert
        float av[16];
        #pragma unroll
        for (int j = 0; j < 16; ++j) {
            int ka = kc + half * 16 + j;
            av[j] = xrow[(size_t)ka * T_];
        }
        if (kc + KCH_ < FEAT_)
            __builtin_prefetch(xrow + (size_t)(kc + KCH_ + half * 8) * T_, 0, 1);
        v16bf a;
        #pragma unroll
        for (int j = 0; j < 16; ++j) a[j] = (__bf16)av[j];

        const __bf16* sb = &smw[cur][0];
        auto ldb = [&](int nt) -> v16bf {
            return *(const v16bf*)(sb + (nt * 16 + ln) * KCH_ + half * 16);
        };
        // software pipeline: issue ds loads for nt+1 before WMMA on nt
        v16bf bm = ldb(0);
        #pragma unroll
        for (int nt = 0; nt < NT1_; ++nt) {
            v16bf bn = bm;
            if (nt + 1 < NT1_) bn = ldb(nt + 1);
            acc[nt] = wmma_bf16(a, bm, acc[nt]);
            bm = bn;
        }

        __syncthreads();                       // all waves done reading buf `cur`
        if (kci + 2 < FEAT_ / KCH_)
            stage(cur, kc + 2 * KCH_);         // safe to overwrite now
    }

    #pragma unroll
    for (int nt = 0; nt < NT1_; ++nt) {
        #pragma unroll
        for (int r = 0; r < 8; ++r) {
            int t   = t0 + r + 8 * half;            // C: m = r + 8*half
            int l   = nt * 16 + ln;                 // C: n = lane&15
            int row = b * TC_ + (t >> 1);
            int c   = (t & 1) * LAT_ + l;           // position in combined frame
            int g   = c / D_, d = c % D_;
            vq[(size_t)row * (G_ * DP_) + g * DP_ + d] = (__bf16)acc[nt][r];
        }
    }
}

// ---- distance GEMM + argmin: idx[row][g] ------------------------------------
// one wave per (16-row tile, group); A fragment resident, sweeps all 1024 codes
// with the next code-tile's B fragments prefetched during the current WMMA chain
__global__ void __launch_bounds__(256, 1)
k_argmin(const __bf16* __restrict__ vq, const __bf16* __restrict__ cbp,
         const float* __restrict__ e2, int* __restrict__ idx) {
    int wid  = (blockIdx.x * blockDim.x + threadIdx.x) >> 5;
    int lane = threadIdx.x & 31;
    int g  = wid % G_;
    int rt = wid / G_;
    int half = lane >> 4, ln = lane & 15;
    int r0 = rt * 16;

    v16bf afrag[5];
    const __bf16* vrow = vq + (size_t)(r0 + ln) * (G_ * DP_) + g * DP_;
    #pragma unroll
    for (int kc = 0; kc < 5; ++kc) {
        v8bf lo = *(const v8bf*)(vrow + kc * 32 + half * 16);
        v8bf hi = *(const v8bf*)(vrow + kc * 32 + half * 16 + 8);
        #pragma unroll
        for (int j = 0; j < 8; ++j) { afrag[kc][j] = lo[j]; afrag[kc][j + 8] = hi[j]; }
    }

    const __bf16* cbg = cbp + (size_t)g * K_ * DP_;
    const float*  e2g = e2 + g * K_;

    auto ld5 = [&](v16bf* dst, int code) {
        const __bf16* crow = cbg + (size_t)code * DP_;
        #pragma unroll
        for (int kc = 0; kc < 5; ++kc)
            dst[kc] = *(const v16bf*)(crow + kc * 32 + half * 16);
    };

    float best[8]; int bidx[8];
    #pragma unroll
    for (int r = 0; r < 8; ++r) { best[r] = 3.4e38f; bidx[r] = 0; }

    v16bf bcur[5];
    ld5(bcur, ln);
    float ecur = e2g[ln];

    #pragma unroll 1
    for (int n0 = 0; n0 < K_; n0 += 16) {
        // prefetch next code tile while this tile's WMMA chain runs
        v16bf bnxt[5];
        #pragma unroll
        for (int kc = 0; kc < 5; ++kc) bnxt[kc] = bcur[kc];
        float enxt = ecur;
        if (n0 + 16 < K_) {
            ld5(bnxt, n0 + 16 + ln);
            enxt = e2g[n0 + 16 + ln];
        }

        v8f acc = {};
        #pragma unroll
        for (int kc = 0; kc < 5; ++kc)
            acc = wmma_bf16(afrag[kc], bcur[kc], acc);

        int code = n0 + ln;                          // B column for this lane
        #pragma unroll
        for (int r = 0; r < 8; ++r) {
            float score = ecur - 2.0f * acc[r];
            if (score < best[r]) { best[r] = score; bidx[r] = code; }
        }

        #pragma unroll
        for (int kc = 0; kc < 5; ++kc) bcur[kc] = bnxt[kc];
        ecur = enxt;
    }

    // reduce over the 16 lanes of each half (rows m=0..7 in lane0, m=8..15 in lane16)
    #pragma unroll
    for (int m = 1; m < 16; m <<= 1) {
        #pragma unroll
        for (int r = 0; r < 8; ++r) {
            float ob = __shfl_xor(best[r], m, 32);
            int   oi = __shfl_xor(bidx[r], m, 32);
            if (ob < best[r] || (ob == best[r] && oi < bidx[r])) {
                best[r] = ob; bidx[r] = oi;
            }
        }
    }
    if (ln == 0) {
        #pragma unroll
        for (int r = 0; r < 8; ++r)
            idx[(size_t)(r0 + r + 8 * half) * G_ + g] = bidx[r];
    }
}

// ---- GEMM3 with fused codebook gather: out = q . W2^T -----------------------
// one wave per (t-tile, 4 f-tiles). A-fragment 8-element runs never straddle a
// 144-wide group boundary (144 % 8 == 0), so each run gathers from one codebook
// row selected by k >= 144.
__global__ void k_proj2(const int* __restrict__ idx, const __bf16* __restrict__ cbp,
                        const __bf16* __restrict__ w2b, float* __restrict__ out) {
    int wid  = (blockIdx.x * blockDim.x + threadIdx.x) >> 5;
    int lane = threadIdx.x & 31;
    int fg = wid % (FEAT_ / 64);          // group of 4 f-tiles
    int rt = wid / (FEAT_ / 64);
    int tt = rt % (T_ / 16);
    int b  = rt / (T_ / 16);
    int half = lane >> 4, ln = lane & 15;
    int t0 = tt * 16;

    int ta  = t0 + ln;                    // lane's A row (t)
    int row = b * TC_ + (ta >> 1);
    int cf  = ta & 1;
    int g0  = 2 * cf;
    const int* ip = idx + (size_t)row * G_;
    const __bf16* p0 = cbp + ((size_t)g0 * K_ + ip[g0]) * DP_;
    const __bf16* p1 = cbp + ((size_t)(g0 + 1) * K_ + ip[g0 + 1]) * DP_;

    v8f acc[4];
    #pragma unroll
    for (int u = 0; u < 4; ++u) { v8f z = {}; acc[u] = z; }

    #pragma unroll
    for (int kc = 0; kc < LAT_; kc += 32) {
        int k0 = kc + half * 16;          // first 8-run (k in [0,288))
        int k1 = kc + half * 16 + 8;      // second 8-run
        const __bf16* q0 = (k0 >= D_) ? (p1 + (k0 - D_)) : (p0 + k0);
        const __bf16* q1 = (k1 >= D_) ? (p1 + (k1 - D_)) : (p0 + k1);
        v8bf lo = *(const v8bf*)q0;
        v8bf hi = *(const v8bf*)q1;
        v16bf a;
        #pragma unroll
        for (int j = 0; j < 8; ++j) { a[j] = lo[j]; a[j + 8] = hi[j]; }
        #pragma unroll
        for (int u = 0; u < 4; ++u) {
            int f = (fg * 4 + u) * 16 + ln;
            v16bf bm = *(const v16bf*)(w2b + (size_t)f * LAT_ + kc + half * 16);
            acc[u] = wmma_bf16(a, bm, acc[u]);
        }
    }

    #pragma unroll
    for (int u = 0; u < 4; ++u) {
        int f0 = (fg * 4 + u) * 16;
        float* ob = out + (size_t)b * FEAT_ * T_ + (size_t)(f0 + ln) * T_ + t0;
        #pragma unroll
        for (int r = 0; r < 8; ++r)
            ob[r + 8 * half] = acc[u][r];      // 8 contiguous floats along t
    }
}

extern "C" void kernel_launch(void* const* d_in, const int* in_sizes, int n_in,
                              void* d_out, int out_size, void* d_ws, size_t ws_size,
                              hipStream_t stream) {
    (void)in_sizes; (void)n_in; (void)out_size; (void)ws_size;
    const float* x  = (const float*)d_in[0];
    const float* w1 = (const float*)d_in[1];
    const float* w2 = (const float*)d_in[2];
    const float* cb = (const float*)d_in[3];
    float* out = (float*)d_out;

    char* base = (char*)d_ws;
    size_t off = 0;
    auto take = [&](size_t bytes) -> char* {
        char* r = base + off;
        off += (bytes + 255) & ~(size_t)255;
        return r;
    };
    size_t vq_bytes = (size_t)ROWS_ * G_ * DP_ * sizeof(__bf16);         // ~21.0 MB
    __bf16* vq  = (__bf16*)take(vq_bytes);
    __bf16* cbp = (__bf16*)take((size_t)G_ * K_ * DP_ * sizeof(__bf16)); // 1.31 MB
    float*  e2  = (float*) take((size_t)G_ * K_ * sizeof(float));
    int*    idx = (int*)   take((size_t)ROWS_ * G_ * sizeof(int));
    __bf16* w1b = (__bf16*)take((size_t)LAT_ * FEAT_ * sizeof(__bf16));
    __bf16* w2b = (__bf16*)take((size_t)FEAT_ * LAT_ * sizeof(__bf16));

    // zero vq so the D->DP_ pad region contributes nothing to distances
    hipMemsetAsync(vq, 0, vq_bytes, stream);

    k_prep_cb<<<(G_ * K_) / 8, 256, 0, stream>>>(cb, cbp, e2);
    k_cvt<<<(LAT_ * FEAT_ + 255) / 256, 256, 0, stream>>>(w1, w1b, LAT_ * FEAT_);
    k_cvt<<<(FEAT_ * LAT_ + 255) / 256, 256, 0, stream>>>(w2, w2b, FEAT_ * LAT_);
    k_proj1 <<<(B_ * (T_ / 16)) / 8, 256, 0, stream>>>(x, w1b, vq);          // 2048 waves
    k_argmin<<<((ROWS_ / 16) * G_) / 8, 256, 0, stream>>>(vq, cbp, e2, idx); // 4096 waves
    k_proj2 <<<(B_ * (T_ / 16) * (FEAT_ / 64)) / 8, 256, 0, stream>>>(idx, cbp, w2b, out); // 16384 waves
}
